// LSTMClassifier_76390288326875
// MI455X (gfx1250) — matrix operand order repaired and checked by reading, not compiled, not run on hardware
//
#include <hip/hip_runtime.h>
#include <hip/hip_bf16.h>

typedef __attribute__((ext_vector_type(16))) _Float16 v16h;
typedef __attribute__((ext_vector_type(8)))  _Float16 v8h;
typedef __attribute__((ext_vector_type(8)))  float    v8f;

constexpr int BATCH = 256;
constexpr int TSTEPS = 512;
constexpr int INSZ  = 64;
constexpr int HID   = 256;
constexpr int HSTR  = HID + 8;   // padded LDS row stride (halves) -> conflict-free banks
constexpr int G4    = 4 * HID;   // 1024 gate pre-activations
constexpr int HEADN = 24 * 5;    // 120

union AFrag { v16h v; v8h g[2]; };

__device__ __forceinline__ float fast_sigmoid(float x) {
    return 1.f / (1.f + __expf(-x));
}
__device__ __forceinline__ float fast_tanh(float x) {
    float e = __expf(-2.f * fabsf(x));
    float r = (1.f - e) / (1.f + e);
    return copysignf(r, x);
}

// ---------------- conversion helpers ----------------
__global__ void cvt_f32_to_f16(const float* __restrict__ s, _Float16* __restrict__ d, int n) {
    for (int i = blockIdx.x * blockDim.x + threadIdx.x; i < n; i += gridDim.x * blockDim.x)
        d[i] = (_Float16)s[i];
}
__global__ void add_bias_f32(const float* __restrict__ a, const float* __restrict__ b,
                             float* __restrict__ d, int n) {
    for (int i = blockIdx.x * blockDim.x + threadIdx.x; i < n; i += gridDim.x * blockDim.x)
        d[i] = a[i] + b[i];
}

// ---------------- fused LSTM layer ----------------
// One workgroup = 16 batch rows, 16 waves; wave w owns hidden cols [w*16, w*16+16),
// i.e. one 16x16 accumulator tile per gate (i,f,g,o) -> pointwise update is
// purely elementwise in registers. Per step:
//   gates[16,1024] = h_{t-1} @ Whh^T + x_t @ Wih^T + bias   (WMMA f16 -> f32)
// The bias lives in pre-splatted C fragments fed to the peeled first k-block's
// WMMA (no per-step accumulator init movs). K-blocks are software-pipelined
// with distance 1 so waits are partial. Weight pointers are laundered per step
// to keep the t-invariant B loads streaming from L2 instead of being hoisted
// into (spilled) registers.
template <int KIN>
__global__ __launch_bounds__(512) void lstm_layer_kernel(
    const _Float16* __restrict__ xin,   // [B, T, KIN] f16
    const _Float16* __restrict__ Wih,   // [1024, KIN] f16 row-major
    const _Float16* __restrict__ Whh,   // [1024, 256] f16 row-major
    const float*    __restrict__ bias,  // [1024] f32 = b_ih + b_hh
    _Float16* __restrict__ out_seq,     // [B, T, 256] f16 or null
    float*    __restrict__ h_final)     // [B, 256] f32 or null
{
    constexpr int KB  = KIN / 32;       // input-side K blocks
    constexpr int NKB = 8 + KB;         // hidden (8) + input K blocks
    __shared__ __align__(16) _Float16 hbuf[16 * HSTR];  // h_{t-1}, padded stride

    const int tid   = threadIdx.x;
    const int lane  = tid & 31;
    const int wave  = tid >> 5;          // 0..15
    const int wg    = blockIdx.x;        // 0..15 -> batch rows [wg*16, wg*16+16)
    const bool hiHalf = (lane >= 16);
    const int l15   = lane & 15;
    const int jbase = wave * 16;         // this wave's hidden-column window

    // zero initial hidden state
    for (int i = tid; i < 16 * HSTR; i += blockDim.x) hbuf[i] = (_Float16)0.f;

    // 4 accumulator tiles, one per gate; per-lane W offsets (t-invariant)
    // bias kept pre-splatted in C fragments, used directly by the first WMMA
    v8f biasFrag[4];
    int boffH[4];   // offset (halves) into Whh for this lane's B fragment
    int boffI[4];   // offset (halves) into Wih
#pragma unroll
    for (int u = 0; u < 4; ++u) {
        const int n = u * HID + jbase + l15;    // W row (gate output column)
        const float bv = bias[n];
#pragma unroll
        for (int r = 0; r < 8; ++r) biasFrag[u][r] = bv;
        boffH[u] = n * HID + (hiHalf ? 8 : 0);
        boffI[u] = n * KIN + (hiHalf ? 8 : 0);
    }

    // persistent cell state fragment (f32)
    v8f cfrag;
#pragma unroll
    for (int r = 0; r < 8; ++r) cfrag[r] = 0.f;

    const int rowg = wg * 16 + l15;      // this lane's A-matrix row (batch index)
    __syncthreads();

#pragma unroll 1
    for (int t = 0; t < TSTEPS; ++t) {
        // launder weight pointers: stops LICM from hoisting t-invariant B loads
        const _Float16* whL = Whh;
        const _Float16* wiL = Wih;
        asm volatile("" : "+s"(whL), "+s"(wiL));

        const size_t xbase = ((size_t)rowg * TSTEPS + t) * KIN + (hiHalf ? 8 : 0);

        auto loadA = [&](int kb, AFrag& a) {
            if (kb < 8) {                       // h_{t-1} from LDS
                const int off = l15 * HSTR + kb * 32 + (hiHalf ? 8 : 0);
                a.g[0] = *(const v8h*)&hbuf[off];
                a.g[1] = *(const v8h*)&hbuf[off + 16];
            } else {                            // x_t from global
                const size_t xb = xbase + (size_t)(kb - 8) * 32;
                a.g[0] = *(const v8h*)&xin[xb];
                a.g[1] = *(const v8h*)&xin[xb + 16];
            }
        };
        auto loadB = [&](int kb, AFrag* bf) {
#pragma unroll
            for (int u = 0; u < 4; ++u) {
                const _Float16* wp = (kb < 8) ? (whL + boffH[u] + kb * 32)
                                              : (wiL + boffI[u] + (kb - 8) * 32);
                bf[u].g[0] = *(const v8h*)wp;
                bf[u].g[1] = *(const v8h*)(wp + 16);
            }
        };

        // ---- software-pipelined GEMM over all K blocks ----
        AFrag a0, a1;
        AFrag b0[4], b1[4];
        loadA(0, a0);
        loadB(0, b0);
        loadA(1, a1);
        loadB(1, b1);

        // peeled kb = 0: C operand comes straight from the persistent bias
        v8f acc[4];
#pragma unroll
        for (int u = 0; u < 4; ++u)
            acc[u] = __builtin_amdgcn_wmma_f32_16x16x32_f16(
                false, a0.v, false, b0[u].v, (short)0, biasFrag[u], false, false);

#pragma unroll
        for (int kb = 1; kb < NKB; ++kb) {
            a0 = a1;
#pragma unroll
            for (int u = 0; u < 4; ++u) b0[u] = b1[u];
            if (kb + 1 < NKB) {
                loadA(kb + 1, a1);
                loadB(kb + 1, b1);
            }
#pragma unroll
            for (int u = 0; u < 4; ++u)
                acc[u] = __builtin_amdgcn_wmma_f32_16x16x32_f16(
                    false, a0.v, false, b0[u].v, (short)0, acc[u], false, false);
        }
        __syncthreads();   // all waves finished reading h_{t-1}; safe to overwrite

        // ---- pointwise LSTM cell update (all 4 gates resident per lane) ----
        {
            const int j = jbase + l15;
#pragma unroll
            for (int r = 0; r < 8; ++r) {
                const float iv = fast_sigmoid(acc[0][r]);
                const float fv = fast_sigmoid(acc[1][r]);
                const float gv = fast_tanh(acc[2][r]);
                const float ov = fast_sigmoid(acc[3][r]);
                const float cn = fv * cfrag[r] + iv * gv;
                cfrag[r] = cn;
                const float hn = ov * fast_tanh(cn);
                const int m = r + (hiHalf ? 8 : 0);              // D-fragment row
                hbuf[m * HSTR + j] = (_Float16)hn;
                if (out_seq)
                    out_seq[((size_t)(wg * 16 + m) * TSTEPS + t) * HID + j] = (_Float16)hn;
                if (h_final && t == TSTEPS - 1)
                    h_final[(size_t)(wg * 16 + m) * HID + j] = hn;
            }
        }
        __syncthreads();   // h_t visible before next step's reads
    }
}

// ---------------- classifier head (tiny: 256x120x256) ----------------
__global__ __launch_bounds__(128) void head_kernel(
    const float* __restrict__ hfin,    // [256, 256]
    const float* __restrict__ Whead,   // [120, 256]
    const float* __restrict__ bhead,   // [120]
    float* __restrict__ out)           // [256, 120]
{
    const int b = blockIdx.x;
    const int o = threadIdx.x;
    if (o >= HEADN) return;
    const float* hv = hfin + (size_t)b * HID;
    const float* wv = Whead + (size_t)o * HID;
    float acc = bhead[o];
#pragma unroll 4
    for (int k = 0; k < HID; ++k) acc += hv[k] * wv[k];
    out[(size_t)b * HEADN + o] = acc;
}

extern "C" void kernel_launch(void* const* d_in, const int* in_sizes, int n_in,
                              void* d_out, int out_size, void* d_ws, size_t ws_size,
                              hipStream_t stream) {
    (void)in_sizes; (void)n_in; (void)out_size; (void)ws_size;
    const float* x     = (const float*)d_in[0];
    const float* Wih0  = (const float*)d_in[1];
    const float* Whh0  = (const float*)d_in[2];
    const float* bih0  = (const float*)d_in[3];
    const float* bhh0  = (const float*)d_in[4];
    const float* Wih1  = (const float*)d_in[5];
    const float* Whh1  = (const float*)d_in[6];
    const float* bih1  = (const float*)d_in[7];
    const float* bhh1  = (const float*)d_in[8];
    const float* Whead = (const float*)d_in[9];
    const float* bhead = (const float*)d_in[10];
    float* out = (float*)d_out;

    // carve workspace (256B-aligned segments)
    char* p = (char*)d_ws;
    auto carve = [&](size_t bytes) -> char* {
        char* r = p;
        p += (bytes + 255) & ~(size_t)255;
        return r;
    };
    _Float16* x16   = (_Float16*)carve((size_t)BATCH * TSTEPS * INSZ * sizeof(_Float16));
    _Float16* Wih0h = (_Float16*)carve((size_t)G4 * INSZ * sizeof(_Float16));
    _Float16* Whh0h = (_Float16*)carve((size_t)G4 * HID * sizeof(_Float16));
    _Float16* Wih1h = (_Float16*)carve((size_t)G4 * HID * sizeof(_Float16));
    _Float16* Whh1h = (_Float16*)carve((size_t)G4 * HID * sizeof(_Float16));
    float*    bias0 = (float*)   carve((size_t)G4 * sizeof(float));
    float*    bias1 = (float*)   carve((size_t)G4 * sizeof(float));
    _Float16* out0  = (_Float16*)carve((size_t)BATCH * TSTEPS * HID * sizeof(_Float16));
    float*    hfin  = (float*)   carve((size_t)BATCH * HID * sizeof(float));

    // 1) precision conversion + bias folding
    cvt_f32_to_f16<<<1024, 256, 0, stream>>>(x,    x16,   BATCH * TSTEPS * INSZ);
    cvt_f32_to_f16<<<256,  256, 0, stream>>>(Wih0, Wih0h, G4 * INSZ);
    cvt_f32_to_f16<<<512,  256, 0, stream>>>(Whh0, Whh0h, G4 * HID);
    cvt_f32_to_f16<<<512,  256, 0, stream>>>(Wih1, Wih1h, G4 * HID);
    cvt_f32_to_f16<<<512,  256, 0, stream>>>(Whh1, Whh1h, G4 * HID);
    add_bias_f32<<<4, 256, 0, stream>>>(bih0, bhh0, bias0, G4);
    add_bias_f32<<<4, 256, 0, stream>>>(bih1, bhh1, bias1, G4);

    // 2) layer 0: writes full output sequence (layer-1 input)
    lstm_layer_kernel<INSZ><<<16, 512, 0, stream>>>(x16, Wih0h, Whh0h, bias0, out0, nullptr);
    // 3) layer 1: only final hidden state needed
    lstm_layer_kernel<HID><<<16, 512, 0, stream>>>(out0, Wih1h, Whh1h, bias1, nullptr, hfin);
    // 4) head
    head_kernel<<<BATCH, 128, 0, stream>>>(hfin, Whead, bhead, out);
}
